// SparseCross2DNA_22668837388654
// MI455X (gfx1250) — matrix operand order repaired and checked by reading, not compiled, not run on hardware
//
#include <hip/hip_runtime.h>
#include <cstdint>
#include <cfloat>

// ---------------- problem constants (from reference) ----------------
#define H_     8
#define DH_    64
#define DIM    512      // H_*DH_
#define FMAP_  48
#define TPF    2304     // FMAP_*FMAP_
#define FCTX   4
#define FQ     4
#define KNUM   9
#define JW     37       // 1 null + FCTX*KNUM
#define XCH    16       // x-positions per window block

// ---------------- WMMA types ----------------
typedef __bf16 bf16;
typedef __attribute__((ext_vector_type(16))) __bf16 v16bf;
typedef __attribute__((ext_vector_type(8)))  __bf16 v8bf;
typedef __attribute__((ext_vector_type(8)))  float  v8f;

// round-to-nearest-even fp32 -> bf16 (low 16 bits)
__device__ __forceinline__ unsigned int bfround(float x) {
  unsigned int u = __builtin_bit_cast(unsigned int, x);
  return (u + 0x7FFFu + ((u >> 16) & 1u)) >> 16;
}
__device__ __forceinline__ unsigned int pack2(float lo, float hi) {
  return (bfround(hi) << 16) | (bfround(lo) & 0xFFFFu);
}
__device__ __forceinline__ bf16 f2bf(float x) {
  return __builtin_bit_cast(bf16, (unsigned short)bfround(x));
}

__device__ __forceinline__ float dot64(const float* a, const float* b) {
  float s = 0.f;
#pragma unroll
  for (int i = 0; i < 16; ++i) {
    float4 av = *(const float4*)(a + i * 4);
    float4 bv = *(const float4*)(b + i * 4);
    s = fmaf(av.x, bv.x, s); s = fmaf(av.y, bv.y, s);
    s = fmaf(av.z, bv.z, s); s = fmaf(av.w, bv.w, s);
  }
  return s;
}

// async 16B global -> LDS copy (ASYNCcnt-tracked) and its drain
__device__ __forceinline__ void async_g2l_b128(unsigned lds_addr, const void* gaddr) {
  asm volatile("global_load_async_to_lds_b128 %0, %1, off"
               :: "v"(lds_addr), "v"(gaddr) : "memory");
}
__device__ __forceinline__ void wait_async0() {
  asm volatile("s_wait_asynccnt 0x0" ::: "memory");
}

// =====================================================================
// fp32 -> bf16 bulk convert (8 elements / thread, 16B stores)
// =====================================================================
__global__ __launch_bounds__(256) void convert_to_bf16(
    const float* __restrict__ src, bf16* __restrict__ dst, long long count)
{
  long long i = ((long long)blockIdx.x * 256 + threadIdx.x) * 8;
  if (i + 8 <= count) {
    float4 a = *(const float4*)(src + i);
    float4 b = *(const float4*)(src + i + 4);
    uint4 pk = make_uint4(pack2(a.x, a.y), pack2(a.z, a.w),
                          pack2(b.x, b.y), pack2(b.z, b.w));
    *(uint4*)(dst + i) = pk;
  } else {
    for (long long j = i; j < count; ++j) dst[j] = f2bf(src[j]);
  }
}

// =====================================================================
// fp32 B[K][N] -> bf16 BT[N][K]  (32x32 tiles via LDS)
// =====================================================================
__global__ __launch_bounds__(256) void transpose_to_bf16(
    const float* __restrict__ B, bf16* __restrict__ BT, int K, int N)
{
  __shared__ float tile[32][33];
  const int tid = threadIdx.x;
  const int k0 = blockIdx.x * 32, n0 = blockIdx.y * 32;
  {
    int r = tid >> 3, c4 = tid & 7;
    float4 v = *(const float4*)(B + (size_t)(k0 + r) * N + n0 + c4 * 4);
    tile[r][c4 * 4 + 0] = v.x; tile[r][c4 * 4 + 1] = v.y;
    tile[r][c4 * 4 + 2] = v.z; tile[r][c4 * 4 + 3] = v.w;
  }
  __syncthreads();
  {
    int nn = tid >> 3, kq = tid & 7;
    uint2 pk = make_uint2(pack2(tile[kq * 4 + 0][nn], tile[kq * 4 + 1][nn]),
                          pack2(tile[kq * 4 + 2][nn], tile[kq * 4 + 3][nn]));
    *(uint2*)(BT + (size_t)(n0 + nn) * K + k0 + kq * 4) = pk;
  }
}

// =====================================================================
// GEMM: C[M,N] = scale * (A[M,K] @ B[K,N]) + bias
// A bf16 row-major [M][K]; B pre-transposed bf16 [N][K]; C fp32.
// Tile 128x128, 8 wave32s; staging via async global->LDS b128 copies,
// double-buffered, 8 WMMAs per wave per K-step.
// Requires: K % 32 == 0, N % 128 == 0.
// =====================================================================
#define BM  128
#define BN  128
#define BK  32
#define LPAD 8

__global__ __launch_bounds__(256) void gemm_bf16_async(
    const bf16* __restrict__ A, const bf16* __restrict__ BT,
    const float* __restrict__ bias, float* __restrict__ C,
    int M, int N, int K, float scale)
{
  __shared__ bf16 smA[2][BM][BK + LPAD];   // [m][k], 16B-aligned rows (80B stride)
  __shared__ bf16 smB[2][BN][BK + LPAD];   // [n][k]

  const int tid  = threadIdx.x;
  const int wave = tid >> 5;
  const int lane = tid & 31;
  const int m0 = blockIdx.x * BM;
  const int n0 = blockIdx.y * BN;

  // issue async copies for one K-slice into buffer `buf`
  auto issue = [&](int k0, int buf) {
#pragma unroll
    for (int p = 0; p < 2; ++p) {          // A: 512 16B chunks
      int idx = tid + p * 256;
      int r   = idx >> 2;
      int c16 = idx & 3;
      int gr  = m0 + r;
      if (gr >= M) gr = M - 1;             // clamp: garbage rows never stored
      async_g2l_b128((unsigned)(uintptr_t)&smA[buf][r][c16 * 8],
                     A + (size_t)gr * K + k0 + c16 * 8);
    }
#pragma unroll
    for (int p = 0; p < 2; ++p) {          // B: 512 16B chunks
      int idx = tid + p * 256;
      int nr  = idx >> 2;
      int c16 = idx & 3;
      async_g2l_b128((unsigned)(uintptr_t)&smB[buf][nr][c16 * 8],
                     BT + (size_t)(n0 + nr) * K + k0 + c16 * 8);
    }
  };

  v8f acc[8] = {};
  const int nsteps = K / BK;

  issue(0, 0);
  wait_async0();

  for (int s = 0; s < nsteps; ++s) {
    __syncthreads();                       // buf[s&1] ready & visible
    const int cur = s & 1;
    if (s + 1 < nsteps) issue((s + 1) * BK, cur ^ 1);

    // A fragment (16x32 bf16): lane<16 -> rows M=lane%16, K 0..7 & 16..23;
    // lane>=16 -> same rows, K 8..15 & 24..31
    const int arow = wave * 16 + (lane & 15);
    const int kb   = (lane < 16) ? 0 : 8;
    v8bf alo = *(const v8bf*)&smA[cur][arow][kb];
    v8bf ahi = *(const v8bf*)&smA[cur][arow][kb + 16];
    v16bf afrag = __builtin_shufflevector(alo, ahi,
        0,1,2,3,4,5,6,7,8,9,10,11,12,13,14,15);

    const int koff = (lane < 16) ? 0 : 8;
#pragma unroll
    for (int nt = 0; nt < 8; ++nt) {
      // B fragment (32x16): lane<16 -> col lane%16, K 0..7 & 16..23; lane>=16 -> K 8..15 & 24..31
      const int bcol = nt * 16 + (lane & 15);
      v8bf blo = *(const v8bf*)&smB[cur][bcol][koff];
      v8bf bhi = *(const v8bf*)&smB[cur][bcol][koff + 16];
      v16bf bfrag = __builtin_shufflevector(blo, bhi,
          0,1,2,3,4,5,6,7,8,9,10,11,12,13,14,15);
      acc[nt] = __builtin_amdgcn_wmma_f32_16x16x32_bf16(
          false, afrag, false, bfrag, (short)0, acc[nt], false, false);
    }

    if (s + 1 < nsteps) wait_async0();     // own copies done before next barrier
  }

  // epilogue: f32 C/D layout -> VGPR r: row r (lanes 0-15) / r+8 (lanes 16-31)
  const int rbase = m0 + wave * 16 + ((lane < 16) ? 0 : 8);
  const int ccol  = lane & 15;
#pragma unroll
  for (int nt = 0; nt < 8; ++nt) {
    int col = n0 + nt * 16 + ccol;
    float bv = bias ? bias[col] : 0.f;
#pragma unroll
    for (int r = 0; r < 8; ++r) {
      int row = rbase + r;
      if (row < M)
        C[(size_t)row * N + col] = acc[nt][r] * scale + bv;
    }
  }
}

// =====================================================================
// BOS attention: one block per (b,h); q[b,pos0] vs [null_k ; k], softmax, @v
// Output written as bf16 (consumed by the bf16 output projection).
// =====================================================================
__global__ __launch_bounds__(256) void bos_attn(
    const float* __restrict__ q, const float* __restrict__ kv,
    const float* __restrict__ null_k, const float* __restrict__ null_v,
    const unsigned char* __restrict__ cmask,
    bf16* __restrict__ out, int n, int cl)
{
  __shared__ float sim[FCTX * TPF + 1];        // 9217
  __shared__ float red[256];
  __shared__ __align__(16) float qsh[DH_];

  const int tid = threadIdx.x;
  const int b = blockIdx.x / H_;
  const int h = blockIdx.x % H_;
  const int J = cl + 1;

  const float* qp = q + (size_t)b * n * DIM + h * DH_;
  if (tid < DH_) qsh[tid] = qp[tid];
  __syncthreads();

  float lmax = -FLT_MAX;
  for (int j = tid; j < J; j += 256) {
    float s;
    if (j == 0) {
      s = dot64(qsh, null_k + h * DH_);
    } else {
      int jj = j - 1;
      s = cmask[b * cl + jj]
            ? dot64(qsh, kv + (size_t)(b * cl + jj) * (2 * DIM) + h * DH_)
            : -FLT_MAX;
    }
    sim[j] = s;
    lmax = fmaxf(lmax, s);
  }
  red[tid] = lmax; __syncthreads();
  for (int st = 128; st > 0; st >>= 1) {
    if (tid < st) red[tid] = fmaxf(red[tid], red[tid + st]);
    __syncthreads();
  }
  const float mx = red[0];
  __syncthreads();

  float lsum = 0.f;
  for (int j = tid; j < J; j += 256) {
    float p = __expf(sim[j] - mx);
    sim[j] = p;
    lsum += p;
  }
  red[tid] = lsum; __syncthreads();
  for (int st = 128; st > 0; st >>= 1) {
    if (tid < st) red[tid] += red[tid + st];
    __syncthreads();
  }
  const float inv = 1.f / red[0];
  __syncthreads();

  const int g = tid >> 6;
  const int d = tid & 63;
  float part = 0.f;
  for (int j = g; j < J; j += 4) {
    float p = sim[j];
    float vd = (j == 0)
        ? null_v[h * DH_ + d]
        : kv[(size_t)(b * cl + j - 1) * (2 * DIM) + DIM + h * DH_ + d];
    part = fmaf(p, vd, part);
  }
  red[tid] = part; __syncthreads();
  if (tid < DH_) {
    float o = red[tid] + red[tid + 64] + red[tid + 128] + red[tid + 192];
    out[(size_t)b * n * DIM + h * DH_ + tid] = f2bf(o * inv);
  }
}

// =====================================================================
// Windowed attention + talking heads (bf16 output).
// One block per (b, f, y, 16-wide x-chunk): sims -> softmax -> W_th mix -> @v
// =====================================================================
__global__ __launch_bounds__(256) void window_attn(
    const float* __restrict__ q, const float* __restrict__ kv,
    const float* __restrict__ null_k, const float* __restrict__ null_v,
    const float* __restrict__ W_th, const unsigned char* __restrict__ cmask,
    bf16* __restrict__ out, int n, int cl)
{
  __shared__ float attn [H_][XCH][JW];
  __shared__ float mixed[H_][XCH][JW];
  __shared__ float wth[H_ * H_];

  const int tid = threadIdx.x;
  const int XT = FMAP_ / XCH;
  int blk = blockIdx.x;
  const int xt = blk % XT;            blk /= XT;
  const int y  = blk % FMAP_;         blk /= FMAP_;
  const int f  = blk % FQ;            blk /= FQ;
  const int b  = blk;
  const int x0 = xt * XCH;

  if (tid < H_ * H_) wth[tid] = W_th[tid];

  const size_t qrow0 =
      (size_t)b * n * DIM + (size_t)(1 + f * TPF + y * FMAP_ + x0) * DIM;

  // ---- Phase A: similarities ----
  const int E = H_ * XCH * JW;   // 4736
  for (int e = tid; e < E; e += 256) {
    int j  = e % JW;
    int xi = (e / JW) % XCH;
    int h  = e / (JW * XCH);
    const float* qp = q + qrow0 + (size_t)xi * DIM + h * DH_;
    float s;
    if (j == 0) {
      s = dot64(qp, null_k + h * DH_);
    } else {
      int c = (j - 1) / KNUM, kk = (j - 1) % KNUM;
      int dy = kk / 3 - 1, dx = kk % 3 - 1;
      int yy = y + dy, xx = x0 + xi + dx;
      if (yy >= 0 && yy < FMAP_ && xx >= 0 && xx < FMAP_) {
        int kidx = c * TPF + yy * FMAP_ + xx;
        s = cmask[b * cl + kidx]
              ? dot64(qp, kv + (size_t)(b * cl + kidx) * (2 * DIM) + h * DH_)
              : -FLT_MAX;
      } else {
        s = -FLT_MAX;   // unfold zero-padding is masked out in the reference
      }
    }
    attn[h][xi][j] = s;
  }
  __syncthreads();

  // ---- Phase B: softmax over j per (h, xi) ----
  for (int r = tid; r < H_ * XCH; r += 256) {
    int h = r / XCH, xi = r % XCH;
    float mx = -FLT_MAX;
    for (int j = 0; j < JW; ++j) mx = fmaxf(mx, attn[h][xi][j]);
    float s = 0.f;
    for (int j = 0; j < JW; ++j) {
      float p = __expf(attn[h][xi][j] - mx);
      attn[h][xi][j] = p;
      s += p;
    }
    float inv = 1.f / s;
    for (int j = 0; j < JW; ++j) attn[h][xi][j] *= inv;
  }
  __syncthreads();

  // ---- Phase C: talking heads ----
  for (int e = tid; e < E; e += 256) {
    int j  = e % JW;
    int xi = (e / JW) % XCH;
    int g  = e / (JW * XCH);
    float m = 0.f;
#pragma unroll
    for (int h = 0; h < H_; ++h) m = fmaf(wth[g * H_ + h], attn[h][xi][j], m);
    mixed[g][xi][j] = m;
  }
  __syncthreads();

  // ---- Phase D: out[g,xi,d] = sum_j mixed * v[g] ----
  const int O = H_ * XCH * DH_;   // 8192
  for (int o = tid; o < O; o += 256) {
    int d  = o % DH_;
    int xi = (o / DH_) % XCH;
    int g  = o / (DH_ * XCH);
    float acc = mixed[g][xi][0] * null_v[g * DH_ + d];
    for (int j = 1; j < JW; ++j) {
      int c = (j - 1) / KNUM, kk = (j - 1) % KNUM;
      int dy = kk / 3 - 1, dx = kk % 3 - 1;
      int yy = y + dy, xx = x0 + xi + dx;
      if (yy >= 0 && yy < FMAP_ && xx >= 0 && xx < FMAP_) {
        int kidx = c * TPF + yy * FMAP_ + xx;
        acc = fmaf(mixed[g][xi][j],
                   kv[(size_t)(b * cl + kidx) * (2 * DIM) + DIM + g * DH_ + d],
                   acc);
      }
    }
    out[qrow0 + (size_t)xi * DIM + g * DH_ + d] = f2bf(acc);
  }
}

// =====================================================================
// launch
// =====================================================================
extern "C" void kernel_launch(void* const* d_in, const int* in_sizes, int n_in,
                              void* d_out, int out_size, void* d_ws, size_t ws_size,
                              hipStream_t stream)
{
  const float*         x       = (const float*)d_in[0];
  const float*         context = (const float*)d_in[1];
  const unsigned char* cmask   = (const unsigned char*)d_in[2];
  const float*         Wq      = (const float*)d_in[3];
  const float*         Wkv     = (const float*)d_in[4];
  const float*         null_k  = (const float*)d_in[5];
  const float*         null_v  = (const float*)d_in[6];
  const float*         W_th    = (const float*)d_in[7];
  const float*         Wo      = (const float*)d_in[8];
  const float*         bo      = (const float*)d_in[9];
  float*               out     = (float*)d_out;

  const int cl = FCTX * TPF;                       // 9216
  const int b  = in_sizes[2] / cl;                 // 2
  const int n  = in_sizes[0] / (b * DIM);          // 9217
  const int bn = b * n, bcl = b * cl;

  // ---- workspace layout ----
  float* q_ws  = (float*)d_ws;                         // [bn][512]   fp32
  float* kv_ws = q_ws + (size_t)bn * DIM;              // [bcl][1024] fp32
  bf16*  xbf   = (bf16*)(kv_ws + (size_t)bcl * 2 * DIM); // [bn][512]
  bf16*  ctxbf = xbf   + (size_t)bn  * DIM;            // [bcl][512]
  bf16*  aobf  = ctxbf + (size_t)bcl * DIM;            // [bn][512]
  bf16*  wqt   = aobf  + (size_t)bn  * DIM;            // [512][512]
  bf16*  wkvt  = wqt   + (size_t)DIM * DIM;            // [1024][512]
  bf16*  wot   = wkvt  + (size_t)2 * DIM * DIM;        // [512][512]

  dim3 blk(256);

  // one-time bf16 conversions / weight transposes
  {
    long long cx = (long long)bn * DIM;
    convert_to_bf16<<<dim3((unsigned)((cx + 2047) / 2048)), blk, 0, stream>>>(x, xbf, cx);
    long long cc = (long long)bcl * DIM;
    convert_to_bf16<<<dim3((unsigned)((cc + 2047) / 2048)), blk, 0, stream>>>(context, ctxbf, cc);
  }
  transpose_to_bf16<<<dim3(DIM / 32, DIM / 32), blk, 0, stream>>>(Wq,  wqt,  DIM, DIM);
  transpose_to_bf16<<<dim3(DIM / 32, (2 * DIM) / 32), blk, 0, stream>>>(Wkv, wkvt, DIM, 2 * DIM);
  transpose_to_bf16<<<dim3(DIM / 32, DIM / 32), blk, 0, stream>>>(Wo,  wot,  DIM, DIM);

  // q = x @ Wq, scaled by DH^-0.5
  gemm_bf16_async<<<dim3((bn + BM - 1) / BM, DIM / BN), blk, 0, stream>>>(
      xbf, wqt, nullptr, q_ws, bn, DIM, DIM, 0.125f);

  // kv = context @ Wkv
  gemm_bf16_async<<<dim3((bcl + BM - 1) / BM, (2 * DIM) / BN), blk, 0, stream>>>(
      ctxbf, wkvt, nullptr, kv_ws, bcl, 2 * DIM, DIM, 1.0f);

  // BOS token attention (fills row 0 per batch, bf16)
  bos_attn<<<dim3(b * H_), blk, 0, stream>>>(
      q_ws, kv_ws, null_k, null_v, cmask, aobf, n, cl);

  // windowed attention (fills rows 1..n-1, bf16)
  window_attn<<<dim3(b * FQ * FMAP_ * (FMAP_ / XCH)), blk, 0, stream>>>(
      q_ws, kv_ws, null_k, null_v, W_th, cmask, aobf, n, cl);

  // final projection: out = ao @ Wo + bo
  gemm_bf16_async<<<dim3((bn + BM - 1) / BM, DIM / BN), blk, 0, stream>>>(
      aobf, wot, bo, out, bn, DIM, DIM, 1.0f);
}
